// Attention_10840497455414
// MI455X (gfx1250) — matrix operand order repaired and checked by reading, not compiled, not run on hardware
//
#include <hip/hip_runtime.h>

#define DIMC   768
#define NHEADS 12
#define HDIM   64
#define SEQ    1024
#define BATCH  8
#define MTOT   (BATCH*SEQ)      /* 8192 */
#define QKVO   (3*DIMC)         /* 2304 */
#define SCALE  0.125f           /* 64^-0.5 */

typedef __attribute__((ext_vector_type(16))) __bf16        v16bf;
typedef __attribute__((ext_vector_type(8)))  float         v8f;
typedef __attribute__((ext_vector_type(4)))  unsigned int  uint4v;
typedef __attribute__((ext_vector_type(4)))  float         f4v;

union Frag16 { v16bf v; unsigned short u[16]; uint4v q[2]; };

// native bf16 convert (RNE) -> single packed cvt on gfx1250
__device__ __forceinline__ unsigned short f2bf(float f) {
  __bf16 h = (__bf16)f;
  return __builtin_bit_cast(unsigned short, h);
}

// LDS byte address for async-DMA VDST: low 32 bits of the flat pointer
__device__ __forceinline__ unsigned ldsaddr(const void* p) {
  return (unsigned)(unsigned long long)p;
}

// CDNA5 async global->LDS copy, 16B per lane, GVS mode (SGPR base + VGPR off).
// Tracked by ASYNCcnt (CDNA5 ISA).
__device__ __forceinline__ void async_b128(unsigned lds, unsigned goff,
                                           const void* sbase) {
  asm volatile("global_load_async_to_lds_b128 %0, %1, %2"
               :: "v"(lds), "v"(goff), "s"(sbase) : "memory");
}
__device__ __forceinline__ void wait_async0() {
  asm volatile("s_wait_asynccnt 0x0" ::: "memory");
}

// ---------------------------------------------------------------------------
// Kernel 1: QKV projection.  out[m,o] = sum_k x[m,k] * w_qkv[o,k]
// Writes Q (pre-scaled) and K as bf16 [B,H,N,D]; V pre-TRANSPOSED as
// bf16 [B,H,D,N] so the attention kernel can stage it with straight
// async copies.  Block tile 64(M) x 128(N); wave tile 32x64 (2x4 WMMA).
// ---------------------------------------------------------------------------
__global__ __launch_bounds__(128)
void qkv_gemm(const float* __restrict__ x, const float* __restrict__ w,
              unsigned short* __restrict__ qb, unsigned short* __restrict__ kb,
              unsigned short* __restrict__ vb)
{
  __shared__ unsigned short As[64][40];    // As[m][k]
  __shared__ unsigned short Bs[128][40];   // Bs[n][k] = w[o0+n][k0+k]

  const int t    = threadIdx.x;
  const int lane = t & 31;
  const int wv   = t >> 5;
  const int hi   = lane >> 4;
  const int lm   = lane & 15;
  const int m0   = blockIdx.x * 64;
  const int o0   = blockIdx.y * 128;
  const int wm   = (wv >> 1) * 32;
  const int wn   = (wv & 1) * 64;

  v8f vzero = {0.f,0.f,0.f,0.f,0.f,0.f,0.f,0.f};
  v8f acc[2][4];
  #pragma unroll
  for (int mt = 0; mt < 2; mt++)
    #pragma unroll
    for (int nt = 0; nt < 4; nt++) acc[mt][nt] = vzero;

  const int lr = t >> 1;           // 0..63 : A tile row
  const int lc = (t & 1) * 16;     // 0/16  : A k-chunk

  for (int k0 = 0; k0 < DIMC; k0 += 32) {
    {
      const float* xp = x + (size_t)(m0 + lr) * DIMC + k0 + lc;
      #pragma unroll
      for (int i = 0; i < 16; i += 4) {
        f4v xv = *(const f4v*)(xp + i);
        As[lr][lc + i + 0] = f2bf(xv.x);
        As[lr][lc + i + 1] = f2bf(xv.y);
        As[lr][lc + i + 2] = f2bf(xv.z);
        As[lr][lc + i + 3] = f2bf(xv.w);
      }
    }
    {
      const float* wp = w + (size_t)(o0 + t) * DIMC + k0;
      #pragma unroll
      for (int i = 0; i < 32; i += 4) {
        f4v wvv = *(const f4v*)(wp + i);
        Bs[t][i + 0] = f2bf(wvv.x);
        Bs[t][i + 1] = f2bf(wvv.y);
        Bs[t][i + 2] = f2bf(wvv.z);
        Bs[t][i + 3] = f2bf(wvv.w);
      }
    }
    __syncthreads();

    Frag16 af[2], bf[4];
    #pragma unroll
    for (int mt = 0; mt < 2; mt++) {
      const unsigned short* p = &As[wm + 16*mt + lm][8*hi];
      af[mt].q[0] = *(const uint4v*)p;
      af[mt].q[1] = *(const uint4v*)(p + 16);
    }
    #pragma unroll
    for (int nt = 0; nt < 4; nt++) {
      const unsigned short* p = &Bs[wn + 16*nt + lm][8*hi];
      bf[nt].q[0] = *(const uint4v*)p;
      bf[nt].q[1] = *(const uint4v*)(p + 16);
    }
    #pragma unroll
    for (int mt = 0; mt < 2; mt++)
      #pragma unroll
      for (int nt = 0; nt < 4; nt++)
        acc[mt][nt] = __builtin_amdgcn_wmma_f32_16x16x32_bf16(
            false, af[mt].v, false, bf[nt].v, (short)0, acc[mt][nt], false, false);
    __syncthreads();
  }

  // scatter into Q/K [B,H,N,D] and V [B,H,D,N] bf16
  #pragma unroll
  for (int nt = 0; nt < 4; nt++) {
    int o     = o0 + wn + 16*nt + lm;
    int three = o / DIMC;
    int rem   = o - three * DIMC;
    int h     = rem >> 6, d = rem & 63;
    if (three == 2) {                       // V, transposed layout
      size_t hb = ((size_t)(h * HDIM + d)) * SEQ;
      #pragma unroll
      for (int mt = 0; mt < 2; mt++)
        #pragma unroll
        for (int j = 0; j < 8; j++) {
          int m = m0 + wm + 16*mt + j + 8*hi;
          int b = m >> 10, n = m & 1023;
          vb[(size_t)b * (NHEADS*HDIM*SEQ) + hb + n] = f2bf(acc[mt][nt][j]);
        }
    } else {                                // Q (scaled) or K
      unsigned short* dst = (three == 0) ? qb : kb;
      float mul = (three == 0) ? SCALE : 1.0f;
      size_t hbase = (size_t)h * SEQ * HDIM + d;
      #pragma unroll
      for (int mt = 0; mt < 2; mt++)
        #pragma unroll
        for (int j = 0; j < 8; j++) {
          int m = m0 + wm + 16*mt + j + 8*hi;
          int b = m >> 10, n = m & 1023;
          dst[(size_t)b * (NHEADS*SEQ*HDIM) + hbase + (size_t)n * HDIM] =
              f2bf(acc[mt][nt][j] * mul);
        }
    }
  }
}

// ---------------------------------------------------------------------------
// Kernel 2: flash attention per (b,h).  Block = 4 waves, 64 Q rows.
// KV tiles (64 wide) staged with double-buffered ASYNC global->LDS copies;
// online softmax; 16 WMMA per tile.  Output bf16 [B,N,C].
// ---------------------------------------------------------------------------
__global__ __launch_bounds__(128)
void attn_kernel(const unsigned short* __restrict__ qb,
                 const unsigned short* __restrict__ kb,
                 const unsigned short* __restrict__ vb,
                 unsigned short* __restrict__ ob)
{
  __shared__ unsigned short Kt[2][64][72];   // Kt[buf][kv][d]
  __shared__ unsigned short Vt[2][64][72];   // Vt[buf][d][kv] (V is pre-transposed)
  __shared__ unsigned short Pt[4][16][72];   // per-wave P (C-layout -> A-layout)

  const int t    = threadIdx.x;
  const int lane = t & 31;
  const int wv   = t >> 5;
  const int hi   = lane >> 4;
  const int lm   = lane & 15;
  const int bh   = blockIdx.y;              // b*H + h
  const int q0   = blockIdx.x * 64;
  const int qrow = q0 + wv * 16;
  const size_t base = (size_t)bh * SEQ * HDIM;
  const unsigned short* kbase = kb + base;  // [kv][d]
  const unsigned short* vbase = vb + base;  // [d][kv]

  // Q fragments (rows qrow+lm, D split into 2 k-steps of 32)
  Frag16 qf[2];
  {
    const unsigned short* qp = qb + base + (size_t)(qrow + lm) * HDIM;
    #pragma unroll
    for (int ks = 0; ks < 2; ks++) {
      qf[ks].q[0] = *(const uint4v*)(qp + 32*ks + 8*hi);
      qf[ks].q[1] = *(const uint4v*)(qp + 32*ks + 16 + 8*hi);
    }
  }

  v8f vzero = {0.f,0.f,0.f,0.f,0.f,0.f,0.f,0.f};
  v8f acc[4];
  acc[0] = vzero; acc[1] = vzero; acc[2] = vzero; acc[3] = vzero;
  float mrow[8], lrow[8];
  #pragma unroll
  for (int j = 0; j < 8; j++) { mrow[j] = -1.0e30f; lrow[j] = 0.0f; }

  const int rr = t >> 1;          // 0..63 : tile row (kv for K, d for V)
  const int c0 = (t & 1) * 32;    // 0/32  : 32-element chunk

  // issue one KV tile's async copies into buffer pb
  auto issue_tile = [&](int pb, int kv0) {
    unsigned kgo = (unsigned)(((kv0 + rr) * HDIM + c0) * 2);
    unsigned klo = ldsaddr(&Kt[pb][rr][c0]);
    async_b128(klo,      kgo,      kbase);
    async_b128(klo + 16, kgo + 16, kbase);
    async_b128(klo + 32, kgo + 32, kbase);
    async_b128(klo + 48, kgo + 48, kbase);
    unsigned vgo = (unsigned)((rr * SEQ + kv0 + c0) * 2);
    unsigned vlo = ldsaddr(&Vt[pb][rr][c0]);
    async_b128(vlo,      vgo,      vbase);
    async_b128(vlo + 16, vgo + 16, vbase);
    async_b128(vlo + 32, vgo + 32, vbase);
    async_b128(vlo + 48, vgo + 48, vbase);
  };

  int pb = 0;
  issue_tile(0, 0);

  for (int kv0 = 0; kv0 < SEQ; kv0 += 64) {
    wait_async0();          // this wave's DMA for tile kv0 done
    __syncthreads();        // everyone's DMA visible in LDS
    if (kv0 + 64 < SEQ) issue_tile(pb ^ 1, kv0 + 64);   // prefetch next tile

    // S = Q K^T   (four 16-col subtiles)
    v8f s[4];
    s[0] = vzero; s[1] = vzero; s[2] = vzero; s[3] = vzero;
    #pragma unroll
    for (int nt = 0; nt < 4; nt++)
      #pragma unroll
      for (int ks = 0; ks < 2; ks++) {
        Frag16 kf;
        const unsigned short* p = &Kt[pb][16*nt + lm][32*ks + 8*hi];
        kf.q[0] = *(const uint4v*)p;
        kf.q[1] = *(const uint4v*)(p + 16);
        s[nt] = __builtin_amdgcn_wmma_f32_16x16x32_bf16(
            false, qf[ks].v, false, kf.v, (short)0, s[nt], false, false);
      }

    // online softmax (row stats reduced across 16-lane half-groups)
    #pragma unroll
    for (int j = 0; j < 8; j++) {
      float mx = fmaxf(fmaxf(s[0][j], s[1][j]), fmaxf(s[2][j], s[3][j]));
      mx = fmaxf(mx, __shfl_xor(mx, 1, 32));
      mx = fmaxf(mx, __shfl_xor(mx, 2, 32));
      mx = fmaxf(mx, __shfl_xor(mx, 4, 32));
      mx = fmaxf(mx, __shfl_xor(mx, 8, 32));
      float mnew  = fmaxf(mrow[j], mx);
      float alpha = __expf(mrow[j] - mnew);
      mrow[j] = mnew;
      float rs = 0.0f;
      #pragma unroll
      for (int nt = 0; nt < 4; nt++) {
        float p = __expf(s[nt][j] - mnew);
        rs += p;
        Pt[wv][j + 8*hi][16*nt + lm] = f2bf(p);   // C-layout -> LDS
      }
      rs += __shfl_xor(rs, 1, 32);
      rs += __shfl_xor(rs, 2, 32);
      rs += __shfl_xor(rs, 4, 32);
      rs += __shfl_xor(rs, 8, 32);
      lrow[j] = lrow[j] * alpha + rs;
      #pragma unroll
      for (int dt = 0; dt < 4; dt++) acc[dt][j] *= alpha;
    }

    // P back as A-fragments (LDS ops are in-order within a wave)
    Frag16 pf[2];
    #pragma unroll
    for (int ks2 = 0; ks2 < 2; ks2++) {
      const unsigned short* pp = &Pt[wv][lm][32*ks2 + 8*hi];
      pf[ks2].q[0] = *(const uint4v*)pp;
      pf[ks2].q[1] = *(const uint4v*)(pp + 16);
    }

    // acc += P * V   (4 d-subtiles x 2 k-steps)
    #pragma unroll
    for (int dt = 0; dt < 4; dt++)
      #pragma unroll
      for (int ks2 = 0; ks2 < 2; ks2++) {
        Frag16 vf;
        const unsigned short* vp2 = &Vt[pb][16*dt + lm][32*ks2 + 8*hi];
        vf.q[0] = *(const uint4v*)vp2;
        vf.q[1] = *(const uint4v*)(vp2 + 16);
        acc[dt] = __builtin_amdgcn_wmma_f32_16x16x32_bf16(
            false, pf[ks2].v, false, vf.v, (short)0, acc[dt], false, false);
      }
    __syncthreads();        // all reads of buffer pb done
    pb ^= 1;
  }

  // normalize + write bf16 [B,N,C]
  const int b = bh / NHEADS, h = bh - b * NHEADS;
  #pragma unroll
  for (int j = 0; j < 8; j++) {
    float inv = 1.0f / lrow[j];
    int n = qrow + j + 8*hi;
    size_t rowbase = ((size_t)b * SEQ + n) * DIMC + h * HDIM;
    #pragma unroll
    for (int dt = 0; dt < 4; dt++)
      ob[rowbase + 16*dt + lm] = f2bf(acc[dt][j] * inv);
  }
}

// ---------------------------------------------------------------------------
// Kernel 3: output projection + bias.  out[m,o] = sum_k a[m,k]*w_proj[o,k]+b[o]
// Block tile 64x128, wave tile 32x64; bf16 A staged with async copies.
// ---------------------------------------------------------------------------
__global__ __launch_bounds__(128)
void proj_gemm(const unsigned short* __restrict__ a, const float* __restrict__ w,
               const float* __restrict__ bias, float* __restrict__ out)
{
  __shared__ unsigned short As[64][40];
  __shared__ unsigned short Bs[128][40];

  const int t    = threadIdx.x;
  const int lane = t & 31;
  const int wv   = t >> 5;
  const int hi   = lane >> 4;
  const int lm   = lane & 15;
  const int m0   = blockIdx.x * 64;
  const int o0   = blockIdx.y * 128;
  const int wm   = (wv >> 1) * 32;
  const int wn   = (wv & 1) * 64;

  v8f vzero = {0.f,0.f,0.f,0.f,0.f,0.f,0.f,0.f};
  v8f acc[2][4];
  #pragma unroll
  for (int mt = 0; mt < 2; mt++)
    #pragma unroll
    for (int nt = 0; nt < 4; nt++) acc[mt][nt] = vzero;

  const int lr = t >> 1;
  const int lc = (t & 1) * 16;

  for (int k0 = 0; k0 < DIMC; k0 += 32) {
    {
      // async-copy A tile (already bf16 in memory)
      unsigned alo  = ldsaddr(&As[lr][lc]);
      unsigned agof = (unsigned)((((m0 + lr) * DIMC) + k0 + lc) * 2);
      async_b128(alo,      agof,      a);
      async_b128(alo + 16, agof + 16, a);
    }
    {
      const float* wp = w + (size_t)(o0 + t) * DIMC + k0;
      #pragma unroll
      for (int i = 0; i < 32; i += 4) {
        f4v wvv = *(const f4v*)(wp + i);
        Bs[t][i + 0] = f2bf(wvv.x);
        Bs[t][i + 1] = f2bf(wvv.y);
        Bs[t][i + 2] = f2bf(wvv.z);
        Bs[t][i + 3] = f2bf(wvv.w);
      }
    }
    wait_async0();
    __syncthreads();

    Frag16 af[2], bf[4];
    #pragma unroll
    for (int mt = 0; mt < 2; mt++) {
      const unsigned short* p = &As[wm + 16*mt + lm][8*hi];
      af[mt].q[0] = *(const uint4v*)p;
      af[mt].q[1] = *(const uint4v*)(p + 16);
    }
    #pragma unroll
    for (int nt = 0; nt < 4; nt++) {
      const unsigned short* p = &Bs[wn + 16*nt + lm][8*hi];
      bf[nt].q[0] = *(const uint4v*)p;
      bf[nt].q[1] = *(const uint4v*)(p + 16);
    }
    #pragma unroll
    for (int mt = 0; mt < 2; mt++)
      #pragma unroll
      for (int nt = 0; nt < 4; nt++)
        acc[mt][nt] = __builtin_amdgcn_wmma_f32_16x16x32_bf16(
            false, af[mt].v, false, bf[nt].v, (short)0, acc[mt][nt], false, false);
    __syncthreads();
  }

  #pragma unroll
  for (int nt = 0; nt < 4; nt++) {
    int o = o0 + wn + 16*nt + lm;
    float bo = bias[o];
    #pragma unroll
    for (int mt = 0; mt < 2; mt++)
      #pragma unroll
      for (int j = 0; j < 8; j++) {
        int m = m0 + wm + 16*mt + j + 8*hi;
        out[(size_t)m * DIMC + o] = acc[mt][nt][j] + bo;
      }
  }
}

// ---------------------------------------------------------------------------
extern "C" void kernel_launch(void* const* d_in, const int* in_sizes, int n_in,
                              void* d_out, int out_size, void* d_ws, size_t ws_size,
                              hipStream_t stream) {
  const float* x      = (const float*)d_in[0];
  const float* w_qkv  = (const float*)d_in[1];
  const float* w_proj = (const float*)d_in[2];
  const float* b_proj = (const float*)d_in[3];

  const size_t HND = (size_t)BATCH * NHEADS * SEQ * HDIM;   // 6,291,456 elems
  unsigned short* ws = (unsigned short*)d_ws;
  unsigned short* qb = ws;              // [B,H,N,D]
  unsigned short* kb = ws + HND;        // [B,H,N,D]
  unsigned short* vb = ws + 2 * HND;    // [B,H,D,N]  (pre-transposed)
  unsigned short* ob = ws + 3 * HND;    // [B,N,C]

  qkv_gemm   <<<dim3(MTOT/64, QKVO/128), 128, 0, stream>>>(x, w_qkv, qb, kb, vb);
  attn_kernel<<<dim3(SEQ/64, BATCH*NHEADS), 128, 0, stream>>>(qb, kb, vb, ob);
  proj_gemm  <<<dim3(MTOT/64, DIMC/128), 128, 0, stream>>>(ob, w_proj, b_proj,
                                                           (float*)d_out);
}
